// SemanticFoldingEncoder_81853486727557
// MI455X (gfx1250) — matrix operand order repaired and checked by reading, not compiled, not run on hardware
//
#include <hip/hip_runtime.h>
#include <hip/hip_bf16.h>
#include <math.h>

// ---------------- problem constants (from reference) ----------------
#define BT     8192      // B*T
#define DIM    512       // D
#define NH     4         // heads
#define DHD    128       // D/H
#define WIN    11        // 2C+1
#define CRAD   5         // C
#define GHW    45
#define GRID2  2025      // GH*GW
#define GPAD   2048      // padded grid row stride (also padded K for fold GEMM)
#define SOUT   2048      // S
#define KTOP   40
#define DFF    1024      // 2*D

typedef __attribute__((ext_vector_type(16))) _Float16 v16h;
typedef __attribute__((ext_vector_type(8)))  _Float16 v8h;
typedef __attribute__((ext_vector_type(8)))  float    v8f;
typedef __attribute__((ext_vector_type(4)))  float    v4f;
typedef int v4i __attribute__((__vector_size__(4 * sizeof(int))));

// ------------- weight convert: f32 (K x N) -> f16 transposed+padded (Npad x Kpad) -------------
__global__ void cvt_w_t(const float* __restrict__ w, _Float16* __restrict__ wt,
                        int K, int N, int Kpad, int Npad) {
    size_t i = (size_t)blockIdx.x * blockDim.x + threadIdx.x;
    size_t total = (size_t)Npad * Kpad;
    if (i >= total) return;
    int k = (int)(i % Kpad);
    int n = (int)(i / Kpad);
    float v = (k < K && n < N) ? w[(size_t)k * N + n] : 0.0f;
    wt[i] = (_Float16)v;
}

// ------------- bias pad: f32 (n) -> f32 (npad), zero tail -------------
__global__ void pad_bias(const float* __restrict__ src, float* __restrict__ dst, int n, int npad) {
    int i = blockIdx.x * blockDim.x + threadIdx.x;
    if (i < npad) dst[i] = (i < n) ? src[i] : 0.0f;
}

// ---------------- embedding gather + LayerNorm ----------------
// one block (256 threads) per token row; D=512 -> 2 elements per thread
__global__ void embed_ln(const int* __restrict__ tokens,
                         const float* __restrict__ table,
                         const float* __restrict__ g,
                         const float* __restrict__ bvec,
                         float* __restrict__ x,          // (BT, D) f32 (residual)
                         _Float16* __restrict__ xn) {    // (BT, D) f16 (GEMM input)
    int row = blockIdx.x;
    int tok = tokens[row];
    const float* e = table + (size_t)tok * DIM;
    int d0 = threadIdx.x, d1 = threadIdx.x + 256;
    float v0 = e[d0], v1 = e[d1];
    float* xr = x + (size_t)row * DIM;
    xr[d0] = v0; xr[d1] = v1;

    __shared__ float red[256];
    red[threadIdx.x] = v0 + v1;
    __syncthreads();
    for (int s = 128; s > 0; s >>= 1) {
        if (threadIdx.x < s) red[threadIdx.x] += red[threadIdx.x + s];
        __syncthreads();
    }
    float mean = red[0] * (1.0f / DIM);
    __syncthreads();
    float c0 = v0 - mean, c1 = v1 - mean;
    red[threadIdx.x] = c0 * c0 + c1 * c1;
    __syncthreads();
    for (int s = 128; s > 0; s >>= 1) {
        if (threadIdx.x < s) red[threadIdx.x] += red[threadIdx.x + s];
        __syncthreads();
    }
    float rinv = rsqrtf(red[0] * (1.0f / DIM) + 1e-5f);
    _Float16* xo = xn + (size_t)row * DIM;
    xo[d0] = (_Float16)(c0 * rinv * g[d0] + bvec[d0]);
    xo[d1] = (_Float16)(c1 * rinv * g[d1] + bvec[d1]);
}

// -------- WMMA GEMM: C(MxN) = act(A(MxK f16, row-major) @ Wt(NxK f16)^T + bias) --------
// All dims pre-padded: M%16==0, N%64==0, K%32==0 -> branch-free inner loop.
// One 16x64 tile per wave: 1 shared A fragment, 4 B fragments, 4 WMMAs per K-step.
// Contiguous k order chosen here (not the ISA table order): lane(half,lm) A frag = A[m0+lm][k0+16h .. +15]
//                       B frag = Wt[n][k0+16h .. +15]  (16 contiguous f16 = 2x b128)
template <int ACT, bool OUT_F16>
__global__ void __launch_bounds__(128)
gemm_wmma(const _Float16* __restrict__ A,
          const _Float16* __restrict__ Wt,     // N x Kd (transposed weights)
          const float* __restrict__ bias,      // N floats (padded) or nullptr
          void* __restrict__ Cout,             // M x N
          int M, int N, int Kd) {
    const int lane = threadIdx.x;               // 0..31 (wave32)
    const int half = lane >> 4;                 // 0 or 1
    const int lm   = lane & 15;
    const int m0 = blockIdx.x * 16;
    const int n0 = (blockIdx.y * blockDim.y + threadIdx.y) * 64;
    if (m0 >= M || n0 >= N) return;             // wave-uniform

    const _Float16* Arow = A  + (size_t)(m0 + lm) * Kd;
    const _Float16* B0   = Wt + (size_t)(n0 + lm) * Kd;
    const _Float16* B1   = B0 + (size_t)16 * Kd;
    const _Float16* B2   = B0 + (size_t)32 * Kd;
    const _Float16* B3   = B0 + (size_t)48 * Kd;

    v8f acc0 = {}, acc1 = {}, acc2 = {}, acc3 = {};
#pragma unroll 2
    for (int k0 = 0; k0 < Kd; k0 += 32) {
        v8h alo = *(const v8h*)(Arow + k0 + 16 * half);
        v8h ahi = *(const v8h*)(Arow + k0 + 16 * half + 8);
        v16h a = __builtin_shufflevector(alo, ahi,
                                         0, 1, 2, 3, 4, 5, 6, 7,
                                         8, 9, 10, 11, 12, 13, 14, 15);
        const int kb = k0 + 16 * half;
        v16h b0v = *(const v16h*)(B0 + kb);
        v16h b1v = *(const v16h*)(B1 + kb);
        v16h b2v = *(const v16h*)(B2 + kb);
        v16h b3v = *(const v16h*)(B3 + kb);
        if (k0 + 64 < Kd)                       // emits global_prefetch_b8
            __builtin_prefetch(Arow + k0 + 64, 0, 1);
        acc0 = __builtin_amdgcn_wmma_f32_16x16x32_f16(false, a, false, b0v, (short)0, acc0, false, false);
        acc1 = __builtin_amdgcn_wmma_f32_16x16x32_f16(false, a, false, b1v, (short)0, acc1, false, false);
        acc2 = __builtin_amdgcn_wmma_f32_16x16x32_f16(false, a, false, b2v, (short)0, acc2, false, false);
        acc3 = __builtin_amdgcn_wmma_f32_16x16x32_f16(false, a, false, b3v, (short)0, acc3, false, false);
    }

    // epilogue: lane holds rows (r + 8*half), cols n0 + s*16 + lm
#pragma unroll
    for (int s = 0; s < 4; ++s) {
        v8f av = (s == 0) ? acc0 : (s == 1) ? acc1 : (s == 2) ? acc2 : acc3;
        int n = n0 + s * 16 + lm;
        float bv = bias ? bias[n] : 0.0f;
#pragma unroll
        for (int r = 0; r < 8; ++r) {
            int m = m0 + r + 8 * half;
            float v = av[r] + bv;
            if (ACT == 1) v = 0.5f * v * (1.0f + erff(v * 0.70710678118f)); // exact gelu
            if (ACT == 2) v = 1.0f / (1.0f + __expf(-v));                   // sigmoid
            if (OUT_F16) ((_Float16*)Cout)[(size_t)m * N + n] = (_Float16)v;
            else         ((float*)Cout)[(size_t)m * N + n]   = v;
        }
    }
}

// ---------------- banded (window-11) attention, one thread per (b,t,h) ----------------
__global__ void banded_attn(const _Float16* __restrict__ q,
                            const _Float16* __restrict__ k,
                            const _Float16* __restrict__ v,
                            _Float16* __restrict__ out) {
    int idx = blockIdx.x * blockDim.x + threadIdx.x;   // BT * NH
    if (idx >= BT * NH) return;
    int h = idx & (NH - 1);
    int t = (idx / NH) & 2047;                         // T = 2048
    int bT = (idx / NH) - t;                           // b*T
    const size_t base = (size_t)(bT + t) * DIM + h * DHD;

    float sc[WIN];
    float mx = -1e30f;
    const _Float16* qp = q + base;
#pragma unroll
    for (int w = 0; w < WIN; ++w) {
        int tt = t + w - CRAD;
        float s = -1e9f;
        if (tt >= 0 && tt < 2048) {
            const _Float16* kp = k + (size_t)(bT + tt) * DIM + h * DHD;
            float acc = 0.0f;
            for (int d = 0; d < DHD; d += 8) {
                v8h qv = *(const v8h*)(qp + d);
                v8h kv = *(const v8h*)(kp + d);
#pragma unroll
                for (int e = 0; e < 8; ++e) acc += (float)qv[e] * (float)kv[e];
            }
            s = acc * 0.08838834764f;                  // 1/sqrt(128)
        }
        sc[w] = s;
        mx = fmaxf(mx, s);
    }
    float den = 0.0f;
#pragma unroll
    for (int w = 0; w < WIN; ++w) { sc[w] = __expf(sc[w] - mx); den += sc[w]; }
    float inv = 1.0f / den;

    _Float16* op = out + base;
    for (int d = 0; d < DHD; d += 8) {
        float accv[8] = {0, 0, 0, 0, 0, 0, 0, 0};
#pragma unroll
        for (int w = 0; w < WIN; ++w) {
            int tt = t + w - CRAD;
            if (tt < 0 || tt >= 2048) continue;
            v8h vv = *(const v8h*)(v + (size_t)(bT + tt) * DIM + h * DHD + d);
#pragma unroll
            for (int e = 0; e < 8; ++e) accv[e] += sc[w] * (float)vv[e];
        }
        v8h ov;
#pragma unroll
        for (int e = 0; e < 8; ++e) ov[e] = (_Float16)(accv[e] * inv);
        *(v8h*)(op + d) = ov;
    }
}

// ---------------- gated residual: ctx = x + sigmoid_gate * oproj ----------------
__global__ void residual_gate(const float* __restrict__ x,
                              const float* __restrict__ gate,
                              const float* __restrict__ oproj,
                              _Float16* __restrict__ ctx, size_t n) {
    size_t i = (size_t)blockIdx.x * blockDim.x + threadIdx.x;
    if (i < n) ctx[i] = (_Float16)(x[i] + gate[i] * oproj[i]);
}

// ------- 5x5 gaussian blur (SAME, zero pad) on 45x45 per image; row stride GPAD -------
__global__ void gauss_smooth(const _Float16* __restrict__ gin, _Float16* __restrict__ sm) {
    const float G1[5] = {0.13533528f, 0.60653066f, 1.0f, 0.60653066f, 0.13533528f};
    const float GNORM = 0.16210282f;                   // 1 / (sum(G1))^2
    size_t i = (size_t)blockIdx.x * blockDim.x + threadIdx.x;
    const size_t total = (size_t)BT * GRID2;
    if (i >= total) return;
    int gx = (int)(i % GHW);
    int gy = (int)((i / GHW) % GHW);
    size_t img = i / GRID2;
    const _Float16* p = gin + img * GPAD;
    float acc = 0.0f;
#pragma unroll
    for (int dy = -2; dy <= 2; ++dy) {
        int yy = gy + dy;
        if (yy < 0 || yy >= GHW) continue;
#pragma unroll
        for (int dx = -2; dx <= 2; ++dx) {
            int xx = gx + dx;
            if (xx < 0 || xx >= GHW) continue;
            acc += G1[dy + 2] * G1[dx + 2] * (float)p[yy * GHW + xx];
        }
    }
    sm[img * GPAD + gy * GHW + gx] = (_Float16)(acc * GNORM);
}

// ---------------- top-40 one-hot mask, one block (256 thr) per row of 2048 ----------
// Row staged into LDS via async global->LDS b128 copies (ASYNCcnt) when the
// toolchain exposes the gfx1250 builtin; overlaps with the sdr zero-fill stores.
__global__ void topk_mask(const float* __restrict__ scores, float* __restrict__ sdr) {
    __shared__ float vals[SOUT];
    __shared__ float rv[256];
    __shared__ int   ri[256];
    int row = blockIdx.x;
    const float* sp = scores + (size_t)row * SOUT;
    float* op = sdr + (size_t)row * SOUT;

    int base = threadIdx.x * 8;                        // 256 * 8 = 2048, 32B aligned
#if __has_builtin(__builtin_amdgcn_global_load_async_to_lds_b128)
    __builtin_amdgcn_global_load_async_to_lds_b128(
        (v4i*)(sp + base),     (v4i*)&vals[base],     0, 0);
    __builtin_amdgcn_global_load_async_to_lds_b128(
        (v4i*)(sp + base + 4), (v4i*)&vals[base + 4], 0, 0);
#else
    *(v4f*)&vals[base]     = *(const v4f*)(sp + base);
    *(v4f*)&vals[base + 4] = *(const v4f*)(sp + base + 4);
#endif
    v4f z = {0.0f, 0.0f, 0.0f, 0.0f};
    *(v4f*)(op + base)     = z;                        // overlaps with async loads
    *(v4f*)(op + base + 4) = z;
#if __has_builtin(__builtin_amdgcn_global_load_async_to_lds_b128)
#if __has_builtin(__builtin_amdgcn_s_wait_asynccnt)
    __builtin_amdgcn_s_wait_asynccnt(0);
#else
    asm volatile("s_wait_asynccnt 0x0" ::: "memory");
#endif
#endif
    __syncthreads();

    for (int it = 0; it < KTOP; ++it) {
        float best = -1e38f; int bi = SOUT;
        for (int i = threadIdx.x; i < SOUT; i += 256) {
            float v = vals[i];
            if (v > best || (v == best && i < bi)) { best = v; bi = i; }
        }
        rv[threadIdx.x] = best; ri[threadIdx.x] = bi;
        __syncthreads();
        for (int s = 128; s > 0; s >>= 1) {
            if (threadIdx.x < s) {
                if (rv[threadIdx.x + s] > rv[threadIdx.x] ||
                    (rv[threadIdx.x + s] == rv[threadIdx.x] && ri[threadIdx.x + s] < ri[threadIdx.x])) {
                    rv[threadIdx.x] = rv[threadIdx.x + s];
                    ri[threadIdx.x] = ri[threadIdx.x + s];
                }
            }
            __syncthreads();
        }
        if (threadIdx.x == 0) { op[ri[0]] = 1.0f; vals[ri[0]] = -1e38f; }
        __syncthreads();
    }
}

// ------------------------------- host launcher -------------------------------
extern "C" void kernel_launch(void* const* d_in, const int* in_sizes, int n_in,
                              void* d_out, int out_size, void* d_ws, size_t ws_size,
                              hipStream_t stream) {
    (void)in_sizes; (void)n_in; (void)out_size; (void)ws_size;
    const int*   tokens = (const int*)d_in[0];
    const float* embed  = (const float*)d_in[1];
    const float* ln_g   = (const float*)d_in[2];
    const float* ln_b   = (const float*)d_in[3];
    const float* wq = (const float*)d_in[4];  const float* bq = (const float*)d_in[5];
    const float* wk = (const float*)d_in[6];  const float* bk = (const float*)d_in[7];
    const float* wv = (const float*)d_in[8];  const float* bv = (const float*)d_in[9];
    const float* wo = (const float*)d_in[10]; const float* bo = (const float*)d_in[11];
    const float* wg = (const float*)d_in[12]; const float* bg = (const float*)d_in[13];
    const float* w1 = (const float*)d_in[14]; const float* b1 = (const float*)d_in[15];
    const float* w2 = (const float*)d_in[16]; const float* b2 = (const float*)d_in[17];
    const float* wfold = (const float*)d_in[18];

    // linear workspace allocator (~210 MB total)
    char* wsb = (char*)d_ws;
    size_t off = 0;
    auto alloc = [&](size_t bytes) -> void* {
        void* p = wsb + off;
        off += (bytes + 255) & ~(size_t)255;
        return p;
    };
    // transposed (+padded) f16 weights: layout N x K
    _Float16* wqT = (_Float16*)alloc((size_t)DIM * DIM * 2);
    _Float16* wkT = (_Float16*)alloc((size_t)DIM * DIM * 2);
    _Float16* wvT = (_Float16*)alloc((size_t)DIM * DIM * 2);
    _Float16* woT = (_Float16*)alloc((size_t)DIM * DIM * 2);
    _Float16* wgT = (_Float16*)alloc((size_t)DIM * DIM * 2);
    _Float16* w1T = (_Float16*)alloc((size_t)DFF * DIM * 2);        // 1024 x 512
    _Float16* w2T = (_Float16*)alloc((size_t)GPAD * DFF * 2);       // 2048(pad 2025) x 1024
    _Float16* wfT = (_Float16*)alloc((size_t)SOUT * GPAD * 2);      // 2048 x 2048(pad 2025)
    float*    b2p = (float*)   alloc((size_t)GPAD * 4);             // padded bias for w2
    float*    xf    = (float*)   alloc((size_t)BT * DIM * 4);
    _Float16* xn16  = (_Float16*)alloc((size_t)BT * DIM * 2);
    _Float16* q16   = (_Float16*)alloc((size_t)BT * DIM * 2);
    _Float16* k16   = (_Float16*)alloc((size_t)BT * DIM * 2);
    _Float16* v16   = (_Float16*)alloc((size_t)BT * DIM * 2);
    float*    gatef = (float*)   alloc((size_t)BT * DIM * 4);
    _Float16* att16 = (_Float16*)alloc((size_t)BT * DIM * 2);
    float*    oprj  = (float*)   alloc((size_t)BT * DIM * 4);
    _Float16* ctx16 = (_Float16*)alloc((size_t)BT * DIM * 2);
    _Float16* h1_16 = (_Float16*)alloc((size_t)BT * DFF * 2);
    _Float16* gr16  = (_Float16*)alloc((size_t)BT * GPAD * 2);      // row stride 2048
    _Float16* sm16  = (_Float16*)alloc((size_t)BT * GPAD * 2);      // row stride 2048

    float* sdr    = (float*)d_out;
    float* scores = (float*)d_out + (size_t)BT * SOUT;

    // 1) weights -> f16 transposed (+padded)
    auto cvtT = [&](const float* s, _Float16* d, int K, int N, int Kp, int Np) {
        size_t tot = (size_t)Np * Kp;
        cvt_w_t<<<(unsigned)((tot + 255) / 256), 256, 0, stream>>>(s, d, K, N, Kp, Np);
    };
    cvtT(wq, wqT, DIM, DIM, DIM, DIM);
    cvtT(wk, wkT, DIM, DIM, DIM, DIM);
    cvtT(wv, wvT, DIM, DIM, DIM, DIM);
    cvtT(wo, woT, DIM, DIM, DIM, DIM);
    cvtT(wg, wgT, DIM, DIM, DIM, DIM);
    cvtT(w1, w1T, DIM, DFF, DIM, DFF);
    cvtT(w2, w2T, DFF, GRID2, DFF, GPAD);       // pad N: 2025 -> 2048
    cvtT(wfold, wfT, GRID2, SOUT, GPAD, SOUT);  // pad K: 2025 -> 2048
    pad_bias<<<(GPAD + 255) / 256, 256, 0, stream>>>(b2, b2p, GRID2, GPAD);

    // 2) embed + LN
    embed_ln<<<BT, 256, 0, stream>>>(tokens, embed, ln_g, ln_b, xf, xn16);

    // GEMM launch: 16x64 tile per wave, 4 waves per block
    dim3 gblk(32, 4);
    auto gdim = [](int M, int N) {
        return dim3((unsigned)(M / 16), (unsigned)((N / 64 + 3) / 4));
    };

    // 3) projections
    gemm_wmma<0, true ><<<gdim(BT, DIM), gblk, 0, stream>>>(xn16, wqT, bq, q16,  BT, DIM, DIM);
    gemm_wmma<0, true ><<<gdim(BT, DIM), gblk, 0, stream>>>(xn16, wkT, bk, k16,  BT, DIM, DIM);
    gemm_wmma<0, true ><<<gdim(BT, DIM), gblk, 0, stream>>>(xn16, wvT, bv, v16,  BT, DIM, DIM);
    gemm_wmma<2, false><<<gdim(BT, DIM), gblk, 0, stream>>>(xn16, wgT, bg, gatef, BT, DIM, DIM);

    // 4) banded attention
    banded_attn<<<(BT * NH + 255) / 256, 256, 0, stream>>>(q16, k16, v16, att16);

    // 5) output projection, gated residual
    gemm_wmma<0, false><<<gdim(BT, DIM), gblk, 0, stream>>>(att16, woT, bo, oprj, BT, DIM, DIM);
    {
        size_t n = (size_t)BT * DIM;
        residual_gate<<<(unsigned)((n + 255) / 256), 256, 0, stream>>>(xf, gatef, oprj, ctx16, n);
    }

    // 6) MLP (w2 output padded to N=2048; pad cols get zero-weight, unused downstream)
    gemm_wmma<1, true><<<gdim(BT, DFF),  gblk, 0, stream>>>(ctx16, w1T, b1,  h1_16, BT, DFF,  DIM);
    gemm_wmma<0, true><<<gdim(BT, GPAD), gblk, 0, stream>>>(h1_16, w2T, b2p, gr16,  BT, GPAD, DFF);

    // 7) gaussian retina smoothing (stride GPAD)
    {
        size_t n = (size_t)BT * GRID2;
        gauss_smooth<<<(unsigned)((n + 255) / 256), 256, 0, stream>>>(gr16, sm16);
    }

    // 8) fold -> scores (K padded to 2048; wfT pad rows are zero so sm16 pad cols are inert)
    gemm_wmma<0, false><<<gdim(BT, SOUT), gblk, 0, stream>>>(sm16, wfT, nullptr, scores, BT, SOUT, GPAD);

    // 9) top-40 one-hot mask (also zero-fills sdr half of d_out)
    topk_mask<<<BT, 256, 0, stream>>>(scores, sdr);
}
